// ConvNet1_2929167696050
// MI455X (gfx1250) — matrix operand order repaired and checked by reading, not compiled, not run on hardware
//
#include <hip/hip_runtime.h>

// ---------------------------------------------------------------------------
// MI455X (gfx1250) fused CNN:
//   conv1(5x5) -> hermval -> maxpool2 : TDM-staged LDS tile + bf16 WMMA
//   conv2(5x5) -> hermval -> maxpool2 : TDM-staged LDS tile + bf16 WMMA
//   flatten -> FC(2704->10)           : bf16 WMMA
// Precision: bf16 inputs/weights, f32 accumulate (V_WMMA_F32_16X16X32_BF16).
// ~8.9 GFLOP vs ~70 MB fused HBM traffic (~3us @ 23.3 TB/s); bf16 K=32 WMMA
// gives ~8x per-issue FLOPs of the f32 K=4 WMMA, balancing compute against
// the memory floor. Conv input tiles are DMA'd into LDS once per block by the
// Tensor Data Mover (tensor_load_to_lds + s_wait_tensorcnt), so the im2col
// gather runs out of LDS instead of re-fetching each element ~25x from L2.
// ---------------------------------------------------------------------------

typedef __bf16 v16bf __attribute__((ext_vector_type(16)));
typedef float  v8f   __attribute__((ext_vector_type(8)));
typedef unsigned int u32x4 __attribute__((ext_vector_type(4)));
typedef int          i32x4 __attribute__((ext_vector_type(4)));
typedef int          i32x8 __attribute__((ext_vector_type(8)));

static __device__ __forceinline__ unsigned short f32_to_bf16(float f) {
  union { float f; unsigned u; } v; v.f = f;
  unsigned u = v.u;
  u += 0x7FFFu + ((u >> 16) & 1u);          // round-to-nearest-even
  return (unsigned short)(u >> 16);
}

// Clenshaw recurrence for degree-4 physicists' Hermite (mirrors reference).
static __device__ __forceinline__ float hermval4(float x, const float c[5]) {
  float x2 = 2.0f * x;
  float c0 = c[3], c1 = c[4], t;
  t = c0; c0 = fmaf(-6.0f, c1, c[2]); c1 = fmaf(c1, x2, t);
  t = c0; c0 = fmaf(-4.0f, c1, c[1]); c1 = fmaf(c1, x2, t);
  t = c0; c0 = fmaf(-2.0f, c1, c[0]); c1 = fmaf(c1, x2, t);
  return fmaf(c1, x2, c0);
}

// ---------------------------------------------------------------------------
// TDM: one 3D-tile DMA load Global->LDS (bf16 elements).
// D# packed per CDNA5 ISA 8.3-8.5:
//   addr = global_addr + 2*(x + y*stride0 + z*stride1), x<tile0,y<tile1,z<tile2
//   OOB reads (y >= tdim1 etc.) return zero (used for conv2's odd row count).
// Issued once per block by wave 0; drained with s_wait_tensorcnt 0.
// ---------------------------------------------------------------------------
static __device__ __forceinline__ void tdm_load_3d_to_lds(
    unsigned lds_off, unsigned long long gaddr,
    int tile0, int tile1, int tile2,
    int tdim0, int tdim1, int tdim2,
    long long stride0, long long stride1)
{
  u32x4 g0;
  g0.x = 1u;                                               // count=1, user desc
  g0.y = lds_off;                                          // LDS byte address
  g0.z = (unsigned)(gaddr & 0xffffffffull);                // global_addr[31:0]
  g0.w = (unsigned)((gaddr >> 32) & 0x1ffffffull)          // global_addr[56:32]
       | (2u << 30);                                       // type = 2 ("image")
  i32x8 g1;
  g1[0] = (1 << 16);                                       // data_size = 2 bytes
  g1[1] = (int)(((unsigned)tdim0 & 0xffffu) << 16);        // tensor_dim0[15:0]
  g1[2] = (int)((((unsigned)tdim0 >> 16) & 0xffffu)
        | (((unsigned)tdim1 & 0xffffu) << 16));            // td0 hi | td1 lo
  g1[3] = (int)((((unsigned)tdim1 >> 16) & 0xffffu)
        | ((unsigned)tile0 << 16));                        // td1 hi | tile_dim0
  g1[4] = (int)((unsigned)tile1 | ((unsigned)tile2 << 16));// tile_dim1 | tile_dim2
  g1[5] = (int)(stride0 & 0xffffffffll);                   // td0_stride[31:0]
  g1[6] = (int)(((stride0 >> 32) & 0xffffll)
        | ((stride1 & 0xffffll) << 16));                   // td0_s hi | td1_s lo
  g1[7] = (int)((stride1 >> 16) & 0xffffffffll);           // td1_stride[47:16]
  i32x4 g2;
  g2[0] = tdim2;                                           // tensor_dim2
  g2[1] = 0;                                               // tensor_dim3
  g2[2] = 0; g2[3] = 0;                                    // td2_stride / tile_dim3 unused (3D tile)
  i32x4 g3 = {0, 0, 0, 0};
  asm volatile("tensor_load_to_lds %0, %1, %2, %3"
               :: "s"(g0), "s"(g1), "s"(g2), "s"(g3) : "memory");
}

// ---------------------------------------------------------------------------
// Elementwise f32 -> bf16 cast (for the network input x).
// ---------------------------------------------------------------------------
__global__ void cast_f32_bf16(const float* __restrict__ in,
                              unsigned short* __restrict__ out, int n) {
  int i = blockIdx.x * blockDim.x + threadIdx.x;
  if (i < n) out[i] = f32_to_bf16(in[i]);
}

// ---------------------------------------------------------------------------
// Pack a [Nreal, Kreal] f32 weight matrix into per-lane bf16 WMMA B fragments:
// out[(s*32 + lane)*16 + j] = W[n = lane%16, k = s*32 + (lane/16)*8 + (j%8) + (j/8)*16],
// zero-padded for k >= Kreal or n >= Nreal.
// ---------------------------------------------------------------------------
__global__ void pack_wmma_b(const float* __restrict__ w,
                            unsigned short* __restrict__ out,
                            int S, int Kreal, int Nreal, int ldk) {
  int idx = blockIdx.x * blockDim.x + threadIdx.x;
  int total = S * 512;
  if (idx >= total) return;
  int j    = idx & 15;
  int lane = (idx >> 4) & 31;
  int s    = idx >> 9;
  int k = s * 32 + (lane >> 4) * 8 + (j & 7) + (j >> 3) * 16;
  int n = lane & 15;
  float v = 0.f;
  if (k < Kreal && n < Nreal) v = w[n * ldk + k];
  out[idx] = f32_to_bf16(v);
}

// ---------------------------------------------------------------------------
// Fused conv(5x5,VALID) + bias + Hermite + maxpool(2,2), implicit GEMM.
// Block = (batch b, group of RPB pooled rows). TDM stages the block's input
// tile [WIN cols x SROWS rows x CIN ch] into LDS; 8 waves each compute a
// 16(M)x16(N) WMMA tile: M = 4 pooled cols x 2x2 window (m = p*4+e),
// N = 16 output channels. K = CIN*25 padded to S*32 (zero packed weights).
// D layout: VGPR i, lanes0-15 -> M=i ; lanes16-31 -> M=8+i, so the 2x2
// maxpool is an in-lane max over acc[0..3] / acc[4..7].
// ---------------------------------------------------------------------------
template<int CIN, int HIN, int WIN, int PH, int PW, int PWT,
         int KREAL, int S, int RPB>
__global__ __launch_bounds__(256)
void conv_herm_pool_wmma(const unsigned short* __restrict__ xin,   // bf16 NCHW
                         const unsigned short* __restrict__ wpack, // packed B
                         const float* __restrict__ bias,
                         const float* __restrict__ coef,
                         unsigned short* __restrict__ outp)        // bf16 [B,16,PH,PW]
{
  constexpr int KTOT  = S * 32;
  constexpr int SROWS = RPB * 2 + 4;          // conv rows (2*RPB) + kernel-1
  constexpr int NSTG  = CIN * SROWS * WIN;
  constexpr int PB    = (PH + RPB - 1) / RPB; // row-groups per image
  __shared__ unsigned short lds_x[NSTG];
  __shared__ int rel[KTOT];

  const int b    = blockIdx.x / PB;
  const int ph0  = (blockIdx.x % PB) * RPB;
  const int row0 = 2 * ph0;

  // im2col LDS-relative offset table: rel[k] = (c*SROWS + kh)*WIN + kw
  for (int k = threadIdx.x; k < KTOT; k += blockDim.x) {
    int kk = (k < KREAL) ? k : (KREAL - 1);
    int c = kk / 25, r = kk % 25;
    rel[k] = (c * SROWS + r / 5) * WIN + (r % 5);
  }

  // Wave 0: one TDM DMA of the input tile; OOB rows (conv2 last pair) -> 0.
  if ((threadIdx.x >> 5) == 0) {
    unsigned lds_off = (unsigned)(unsigned long long)(&lds_x[0]);
    unsigned long long gaddr = (unsigned long long)(const void*)xin
        + 2ull * ((unsigned long long)b * CIN * HIN * WIN
                  + (unsigned long long)row0 * WIN);
    tdm_load_3d_to_lds(lds_off, gaddr,
                       /*tile */ WIN, SROWS, CIN,
                       /*tensor*/ WIN, HIN - row0, CIN,
                       /*strides*/ (long long)WIN, (long long)HIN * WIN);
    __builtin_amdgcn_s_wait_tensorcnt(0);
  }
  __syncthreads();

  const int w    = threadIdx.x >> 5;
  const int lane = threadIdx.x & 31;
  int ph = ph0 + w / PWT;                      // pooled row for this wave
  if (ph > PH - 1) ph = PH - 1;                // dup waves write same bytes
  const int pwt = w % PWT;

  const int mrow  = lane & 15;   // A row M (producer) / D column N (consumer)
  const int khalf = lane >> 4;

  const int p = mrow >> 2, e = mrow & 3;       // pooled-col offset, 2x2 element
  int pw_in = pwt * 4 + p;
  if (pw_in >= PW) pw_in = PW - 1;             // clamp (never stored)
  const int pos = (2 * (ph - ph0) + (e >> 1)) * WIN + 2 * pw_in + (e & 1);

  v8f acc = {};
  for (int s = 0; s < S; ++s) {
    const int kb = s * 32 + khalf * 8;
    // A fragment: element j<8 -> K = kb+j ; j>=8 -> K = kb+16+(j-8)
    union { unsigned short u[16]; v16bf v; } a;
#pragma unroll
    for (int j = 0; j < 8; ++j) {
      a.u[j]     = lds_x[pos + rel[kb + j]];
      a.u[8 + j] = lds_x[pos + rel[kb + 16 + j]];
    }
    union { uint4 q[2]; v16bf v; } bfr;
    const uint4* wp = (const uint4*)(wpack + (((size_t)s * 32 + lane) << 4));
    bfr.q[0] = wp[0]; bfr.q[1] = wp[1];
    acc = __builtin_amdgcn_wmma_f32_16x16x32_bf16(false, a.v, false, bfr.v,
                                                  (short)0, acc, false, false);
  }

  // Epilogue: bias + Hermite + 2x2 maxpool, all in-lane.
  float c[5];
#pragma unroll
  for (int i = 0; i < 5; ++i) c[i] = coef[i];
  const float bn = bias[mrow];
  float h[8];
#pragma unroll
  for (int i = 0; i < 8; ++i) h[i] = hermval4(acc[i] + bn, c);
  const float r0 = fmaxf(fmaxf(h[0], h[1]), fmaxf(h[2], h[3])); // pooled p = khalf*2
  const float r1 = fmaxf(fmaxf(h[4], h[5]), fmaxf(h[6], h[7])); // pooled p = khalf*2+1

  const long obase = (((long)b * 16 + mrow) * PH + ph) * (long)PW;
  const int pw0 = pwt * 4 + khalf * 2;
  if (pw0 < PW)     outp[obase + pw0]     = f32_to_bf16(r0);
  if (pw0 + 1 < PW) outp[obase + pw0 + 1] = f32_to_bf16(r1);
}

// ---------------------------------------------------------------------------
// FC: [512,2704](bf16) x packed [2704,16](bf16, cols 10..15 zero) -> f32 [512,10]
// One wave per 16-row M tile; A rows contiguous -> b128 loads.
// ---------------------------------------------------------------------------
__global__ __launch_bounds__(256)
void fc_wmma(const unsigned short* __restrict__ a,      // bf16 [512,2704] (+slack)
             const unsigned short* __restrict__ wpack,  // S = 85 packed steps
             const float* __restrict__ bias,
             float* __restrict__ out)                   // f32 [512,10]
{
  const int lane  = threadIdx.x & 31;
  const int wave  = blockIdx.x * (blockDim.x >> 5) + (threadIdx.x >> 5);
  const int mrow  = lane & 15;
  const int khalf = lane >> 4;
  const long arow = ((long)(wave * 16 + mrow)) * 2704;

  v8f acc = {};
  for (int s = 0; s < 85; ++s) {
    const int kb = s * 32 + khalf * 8;        // multiple of 8 -> 16B aligned
    union { uint4 q[2]; v16bf v; } af;
    af.q[0] = *(const uint4*)(a + arow + kb);
    af.q[1] = *(const uint4*)(a + arow + kb + 16);
    union { uint4 q[2]; v16bf v; } bfr;
    const uint4* wp = (const uint4*)(wpack + (((size_t)s * 32 + lane) << 4));
    bfr.q[0] = wp[0]; bfr.q[1] = wp[1];
    acc = __builtin_amdgcn_wmma_f32_16x16x32_bf16(false, af.v, false, bfr.v,
                                                  (short)0, acc, false, false);
  }

  if (mrow < 10) {
    const float bn = bias[mrow];
#pragma unroll
    for (int i = 0; i < 8; ++i) {
      const int m = wave * 16 + khalf * 8 + i;  // D: VGPR i -> M = khalf*8 + i
      out[m * 10 + mrow] = acc[i] + bn;
    }
  }
}

// ---------------------------------------------------------------------------
// Host launcher
// ---------------------------------------------------------------------------
extern "C" void kernel_launch(void* const* d_in, const int* in_sizes, int n_in,
                              void* d_out, int out_size, void* d_ws, size_t ws_size,
                              hipStream_t stream) {
  const float* x       = (const float*)d_in[0];  // [512,3,64,64]
  const float* coef    = (const float*)d_in[1];  // [5]
  const float* conv1_w = (const float*)d_in[2];  // [16,3,5,5]
  const float* conv1_b = (const float*)d_in[3];  // [16]
  const float* conv2_w = (const float*)d_in[4];  // [16,16,5,5]
  const float* conv2_b = (const float*)d_in[5];  // [16]
  const float* fc1_w   = (const float*)d_in[6];  // [10,2704]
  const float* fc1_b   = (const float*)d_in[7];  // [10]
  float* out = (float*)d_out;                    // [512,10]

  // Workspace layout (256B aligned), total ~30.2 MB.
  char* ws = (char*)d_ws;
  size_t o = 0;
  auto alloc = [&](size_t bytes) { size_t r = o; o = (o + bytes + 255) & ~(size_t)255; return r; };
  const size_t x_elems  = (size_t)512 * 3 * 64 * 64;         // 6,291,456
  const size_t p1_elems = (size_t)512 * 16 * 30 * 30;        // 7,372,800
  const size_t p2_elems = (size_t)512 * 16 * 13 * 13;        // 1,384,448
  unsigned short* xbf   = (unsigned short*)(ws + alloc(x_elems * 2));
  unsigned short* w1p   = (unsigned short*)(ws + alloc((size_t)3  * 512 * 2));
  unsigned short* w2p   = (unsigned short*)(ws + alloc((size_t)13 * 512 * 2));
  unsigned short* wfp   = (unsigned short*)(ws + alloc((size_t)85 * 512 * 2));
  unsigned short* pool1 = (unsigned short*)(ws + alloc(p1_elems * 2));
  unsigned short* pool2 = (unsigned short*)(ws + alloc(p2_elems * 2 + 256)); // +slack for padded-K b128 reads
  (void)ws_size; (void)in_sizes; (void)n_in; (void)out_size;

  // 1) input to bf16
  cast_f32_bf16<<<(int)((x_elems + 255) / 256), 256, 0, stream>>>(x, xbf, (int)x_elems);
  // 2) pack weight matrices into WMMA B-fragment order
  pack_wmma_b<<<(3  * 512 + 255) / 256, 256, 0, stream>>>(conv1_w, w1p, 3,  75,   16, 75);
  pack_wmma_b<<<(13 * 512 + 255) / 256, 256, 0, stream>>>(conv2_w, w2p, 13, 400,  16, 400);
  pack_wmma_b<<<(85 * 512 + 255) / 256, 256, 0, stream>>>(fc1_w,   wfp, 85, 2704, 10, 2704);

  // 3) conv1: block = (b, pooled row); 8 waves = 8 col-tiles of 4 pooled cols
  conv_herm_pool_wmma<3, 64, 64, 30, 30, 8, 75, 3, 1>
      <<<512 * 30, 256, 0, stream>>>(xbf, w1p, conv1_b, coef, pool1);
  // 4) conv2: block = (b, pooled row pair); 8 waves = 2 rows x 4 col-tiles
  conv_herm_pool_wmma<16, 30, 30, 13, 13, 4, 400, 13, 2>
      <<<512 * 7, 256, 0, stream>>>(pool1, w2p, conv2_b, coef, pool2);
  // 5) FC: 32 m-tiles -> 4 blocks of 8 waves
  fc_wmma<<<4, 256, 0, stream>>>(pool2, wfp, fc1_b, out);
}
